// DecoderStep_25185688224036
// MI455X (gfx1250) — matrix operand order repaired and checked by reading, not compiled, not run on hardware
//
#include <hip/hip_runtime.h>
#include <hip/hip_bf16.h>

// Problem constants (match reference)
constexpr int Bc = 32;
constexpr int Nc = 2000;
constexpr int Ec = 128;
constexpr int Hc = 8;
constexpr int Dc = 16;
constexpr float NEGF = -1e9f;
constexpr float LNEPS = 1e-5f;
constexpr int TILES = Nc / 16;        // 125 node tiles of 16
constexpr int TILES_PER_BLK = 5;
constexpr int TGROUPS = TILES / TILES_PER_BLK;  // 25

typedef __attribute__((ext_vector_type(16))) __bf16 v16bf;
typedef __attribute__((ext_vector_type(8)))  float  v8f;

union BFrag {
    v16bf v;
    unsigned int u[8];
    __bf16 h[16];
};

// LDS offset of a __shared__ object: generic (flat) address of the LDS
// aperture carries the LDS byte offset in addr[31:0] (ISA 10.2 aperture map).
__device__ __forceinline__ unsigned lds_off(const void* p) {
    return (unsigned)(uintptr_t)p;
}

// CDNA5 async DMA: global -> LDS, 16 bytes, no VGPR data path (ASYNCcnt).
__device__ __forceinline__ void async_load_b128_to_lds(unsigned lds_byte_addr,
                                                       const float* gptr) {
    asm volatile("global_load_async_to_lds_b128 %0, %1, off"
                 :
                 : "v"(lds_byte_addr),
                   "v"((unsigned long long)(uintptr_t)gptr)
                 : "memory");
}

// ---------------------------------------------------------------------------
// K1: q = decoder_state @ Wq + bq    [B,E]
// ---------------------------------------------------------------------------
__global__ void qproj_kernel(const float* __restrict__ ds,
                             const float* __restrict__ Wq,
                             const float* __restrict__ bq,
                             float* __restrict__ q) {
    __shared__ float s_ds[Ec];
    const int b = blockIdx.x, e = threadIdx.x;
    s_ds[e] = ds[b * Ec + e];
    __syncthreads();
    float acc = bq[e];
#pragma unroll 8
    for (int i = 0; i < Ec; ++i) acc += s_ds[i] * Wq[i * Ec + e];
    q[b * Ec + e] = acc;
}

// ---------------------------------------------------------------------------
// K2: fused K/V projection (bf16 WMMA, fp32 accum) + masked attention logits.
// Block = 256 threads = 8 wave32s; wave w owns head w's 16-wide column block.
// Node tiles are staged global->LDS with async DMA, double buffered so the
// DMA of tile t+1 overlaps WMMA on tile t. K never leaves LDS; V -> workspace.
// ---------------------------------------------------------------------------
__global__ void __launch_bounds__(256, 1)
kv_att_kernel(const float* __restrict__ ne,
              const float* __restrict__ Wk, const float* __restrict__ bk,
              const float* __restrict__ Wv, const float* __restrict__ bv,
              const int* __restrict__ mask,
              const float* __restrict__ q,
              float* __restrict__ att,        // [B,H,N]
              float* __restrict__ Vout) {     // [B,N,E]
    // fp32 staging tiles (written ONLY by async DMA), padded rows: 132 floats.
    __shared__ alignas(16) float s_stage[2][16][132];
    __shared__ float s_k[16][132];   // K tile fp32
    __shared__ float s_v[16][132];   // V tile fp32
    __shared__ float s_q[Ec];

    const int b    = blockIdx.x;
    const int t0   = blockIdx.y * TILES_PER_BLK;
    const int tid  = threadIdx.x;
    const int w    = tid >> 5;        // wave id == head id == output col block
    const int lane = tid & 31;
    const int hlf  = lane >> 4;       // which 16-lane half
    const int m    = lane & 15;
    const int col  = w * 16 + m;      // output column (0..127) this lane owns
    const int kh   = hlf * 16;        // B-fragment K offset for upper half-wave

    if (tid < Ec) s_q[tid] = q[b * Ec + tid];

    // Preload B fragments (weight columns) once per block: contiguous k order (same as A),
    // lane element e -> (K = 32*step + 16*hlf + e, N = col).
    BFrag fk[4], fv[4];
#pragma unroll
    for (int s = 0; s < 4; ++s) {
#pragma unroll
        for (int e = 0; e < 16; ++e) {
            const int K = 32 * s + kh + e;
            fk[s].h[e] = (__bf16)Wk[K * Ec + col];
            fv[s].h[e] = (__bf16)Wv[K * Ec + col];
        }
    }
    const float bkc = bk[col];
    const float bvc = bv[col];

    // Issue async DMA for the first tile into buffer 0.
    // Each tile = 512 x 16B chunks; each thread issues exactly 2 (ASYNCcnt=2).
    {
        const float* src = ne + ((size_t)b * Nc + t0 * 16) * Ec;
#pragma unroll
        for (int c = tid; c < 512; c += 256) {
            const int row = c >> 5;               // 32 chunks per 128-float row
            const int f0  = (c & 31) * 4;         // float offset within row
            async_load_b128_to_lds(lds_off(&s_stage[0][row][f0]),
                                   src + (size_t)row * Ec + f0);
        }
    }

    for (int tt = 0; tt < TILES_PER_BLK; ++tt) {
        const int tile = t0 + tt;
        const int gn0  = tile * 16;
        const int buf  = tt & 1;

        // All waves done reading the buffer we are about to refill.
        __syncthreads();

        if (tt + 1 < TILES_PER_BLK) {
            const float* src = ne + ((size_t)b * Nc + (gn0 + 16)) * Ec;
#pragma unroll
            for (int c = tid; c < 512; c += 256) {
                const int row = c >> 5;
                const int f0  = (c & 31) * 4;
                async_load_b128_to_lds(lds_off(&s_stage[buf ^ 1][row][f0]),
                                       src + (size_t)row * Ec + f0);
            }
            // current buffer's 2 loads done; next tile's 2 may be in flight
            asm volatile("s_wait_asynccnt 0x2" ::: "memory");
        } else {
            asm volatile("s_wait_asynccnt 0x0" ::: "memory");
        }
        __syncthreads();   // every wave's DMA for `buf` has landed in LDS

        // A-fragments: fp32 pairs from LDS stage -> bf16 (contiguous k = 16*hlf + e, same as B)
        v8f acck = {};
        v8f accv = {};
        const float* rowp = &s_stage[buf][m][0];
#pragma unroll
        for (int s = 0; s < 4; ++s) {
            BFrag a;
            const int kb = 32 * s + 16 * hlf;
#pragma unroll
            for (int p = 0; p < 4; ++p) {
                const float2 f0 = *(const float2*)&rowp[kb + 2 * p];
                const float2 f1 = *(const float2*)&rowp[kb + 8 + 2 * p];
                a.h[2 * p]     = (__bf16)f0.x;
                a.h[2 * p + 1] = (__bf16)f0.y;
                a.h[8 + 2 * p]     = (__bf16)f1.x;
                a.h[8 + 2 * p + 1] = (__bf16)f1.y;
            }
            acck = __builtin_amdgcn_wmma_f32_16x16x32_bf16(
                false, a.v, false, fk[s].v, (short)0, acck, false, false);
            accv = __builtin_amdgcn_wmma_f32_16x16x32_bf16(
                false, a.v, false, fv[s].v, (short)0, accv, false, false);
        }

        // Scatter C (VGPR r -> node = 8*hlf + r, column = col) + bias.
#pragma unroll
        for (int r = 0; r < 8; ++r) {
            s_k[8 * hlf + r][col] = acck[r] + bkc;
            s_v[8 * hlf + r][col] = accv[r] + bvc;
        }
        __syncthreads();

        // Attention logits: 8 heads x 16 nodes, q . k / sqrt(D), masked.
        if (tid < Hc * 16) {
            const int h = tid >> 4, node = tid & 15;
            float a = 0.f;
#pragma unroll
            for (int d = 0; d < Dc; ++d)
                a += s_k[node][h * Dc + d] * s_q[h * Dc + d];
            a *= 0.25f;  // 1/sqrt(16)
            const int gn = gn0 + node;
            if (mask[b * Nc + gn] == 0) a = NEGF;
            att[((size_t)b * Hc + h) * Nc + gn] = a;
        }

        // V tile -> workspace (fp32).
        for (int i = tid; i < 16 * Ec; i += 256) {
            const int node = i >> 7, e = i & 127;
            Vout[((size_t)b * Nc + gn0 + node) * Ec + e] = s_v[node][e];
        }
    }
}

// ---------------------------------------------------------------------------
// K3: per-(b,h) softmax over N + weighted V sum -> ctx[b, h*16 .. h*16+15]
// ---------------------------------------------------------------------------
__global__ void __launch_bounds__(256, 1)
softmax_ctx_kernel(const float* __restrict__ att,
                   const float* __restrict__ V,
                   float* __restrict__ ctx) {
    __shared__ float red[256];
    __shared__ float red2[16][17];
    const int bh = blockIdx.x;
    const int b = bh >> 3, h = bh & 7;
    const float* arow = att + (size_t)bh * Nc;
    const int tid = threadIdx.x;

    float mx = -3.4e38f;
    for (int n = tid; n < Nc; n += 256) mx = fmaxf(mx, arow[n]);
    red[tid] = mx;
    __syncthreads();
    for (int s = 128; s > 0; s >>= 1) {
        if (tid < s) red[tid] = fmaxf(red[tid], red[tid + s]);
        __syncthreads();
    }
    mx = red[0];
    __syncthreads();

    float se = 0.f;
    for (int n = tid; n < Nc; n += 256) se += __expf(arow[n] - mx);
    red[tid] = se;
    __syncthreads();
    for (int s = 128; s > 0; s >>= 1) {
        if (tid < s) red[tid] += red[tid + s];
        __syncthreads();
    }
    const float inv = 1.f / red[0];

    const int d = tid & 15, g = tid >> 4;  // 16 dims x 16 n-groups
    float part = 0.f;
    for (int n = g; n < Nc; n += 16) {
        const float wv = __expf(arow[n] - mx) * inv;
        part += wv * V[((size_t)b * Nc + n) * Ec + h * Dc + d];
    }
    red2[g][d] = part;
    __syncthreads();
    if (tid < 16) {
        float s = 0.f;
#pragma unroll
        for (int gg = 0; gg < 16; ++gg) s += red2[gg][tid];
        ctx[b * Ec + h * Dc + tid] = s;
    }
}

// ---------------------------------------------------------------------------
// K4: ctx @ Wo + bo -> LayerNorm -> @ Wp + bp = logits  (one block per batch)
// ---------------------------------------------------------------------------
__global__ void out_ln_proj_kernel(const float* __restrict__ ctx,
                                   const float* __restrict__ Wo, const float* __restrict__ bo,
                                   const float* __restrict__ Wp, const float* __restrict__ bp,
                                   const float* __restrict__ lng, const float* __restrict__ lnb,
                                   float* __restrict__ logits) {
    __shared__ float s_ctx[Ec];
    __shared__ float s_nrm[Ec];
    __shared__ float red[Ec];
    const int b = blockIdx.x, e = threadIdx.x;
    s_ctx[e] = ctx[b * Ec + e];
    __syncthreads();

    float o = bo[e];
#pragma unroll 8
    for (int i = 0; i < Ec; ++i) o += s_ctx[i] * Wo[i * Ec + e];

    red[e] = o;
    __syncthreads();
    for (int s = 64; s > 0; s >>= 1) {
        if (e < s) red[e] += red[e + s];
        __syncthreads();
    }
    const float mu = red[0] * (1.f / Ec);
    __syncthreads();
    const float dm = o - mu;
    red[e] = dm * dm;
    __syncthreads();
    for (int s = 64; s > 0; s >>= 1) {
        if (e < s) red[e] += red[e + s];
        __syncthreads();
    }
    const float var = red[0] * (1.f / Ec);
    const float nr = dm * rsqrtf(var + LNEPS) * lng[e] + lnb[e];
    s_nrm[e] = nr;
    __syncthreads();

    float lg = bp[e];
#pragma unroll 8
    for (int i = 0; i < Ec; ++i) lg += s_nrm[i] * Wp[i * Ec + e];
    logits[b * Ec + e] = lg;
}

// ---------------------------------------------------------------------------
// K5: scores = logits . NE[b,n] with revisit penalty + shortcut bonus
// ---------------------------------------------------------------------------
__global__ void scores_kernel(const float* __restrict__ logits,
                              const float* __restrict__ ne,
                              const unsigned char* __restrict__ visited,
                              const int* __restrict__ prev_node,
                              const unsigned char* __restrict__ all_visited,
                              const float* __restrict__ ew,
                              const float* __restrict__ sb,
                              const float* __restrict__ rp,
                              float* __restrict__ out) {
    __shared__ float s_lg[Ec];
    const int b = blockIdx.x;
    const int n = blockIdx.y * 256 + threadIdx.x;
    if (threadIdx.x < Ec) s_lg[threadIdx.x] = logits[b * Ec + threadIdx.x];
    __syncthreads();
    if (n >= Nc) return;

    const float* ner = ne + ((size_t)b * Nc + n) * Ec;
    float s = 0.f;
#pragma unroll 8
    for (int e = 0; e < Ec; ++e) s += s_lg[e] * ner[e];

    const bool av = all_visited[b] != 0;
    if (!av && visited[(size_t)b * Nc + n]) s = rp[0];

    const float* ewb = ew + (size_t)b * Nc * Nc;
    const int pv = prev_node[b];
    const float direct = ewb[(size_t)pv * Nc];
    const float via = ewb[(size_t)pv * Nc + n] + ewb[(size_t)n * Nc];
    if (av && via < direct) s += sb[0] * (direct - via) / direct;

    out[(size_t)b * Nc + n] = s;
}

// ---------------------------------------------------------------------------
extern "C" void kernel_launch(void* const* d_in, const int* in_sizes, int n_in,
                              void* d_out, int out_size, void* d_ws, size_t ws_size,
                              hipStream_t stream) {
    const float*         ds      = (const float*)d_in[0];
    const float*         ne      = (const float*)d_in[1];
    const int*           mask    = (const int*)d_in[2];
    const float*         ew      = (const float*)d_in[3];
    const unsigned char* visited = (const unsigned char*)d_in[4];
    const int*           prev    = (const int*)d_in[5];
    const unsigned char* allv    = (const unsigned char*)d_in[6];
    const float* Wq = (const float*)d_in[7];
    const float* bq = (const float*)d_in[8];
    const float* Wk = (const float*)d_in[9];
    const float* bk = (const float*)d_in[10];
    const float* Wv = (const float*)d_in[11];
    const float* bv = (const float*)d_in[12];
    const float* Wo = (const float*)d_in[13];
    const float* bo = (const float*)d_in[14];
    const float* Wp = (const float*)d_in[15];
    const float* bp = (const float*)d_in[16];
    const float* lng = (const float*)d_in[17];
    const float* lnb = (const float*)d_in[18];
    const float* sb  = (const float*)d_in[19];
    const float* rp  = (const float*)d_in[20];
    float* out = (float*)d_out;

    float* ws        = (float*)d_ws;
    float* ws_q      = ws;                                   // B*E
    float* ws_att    = ws_q + Bc * Ec;                       // B*H*N
    float* ws_V      = ws_att + (size_t)Bc * Hc * Nc;        // B*N*E
    float* ws_ctx    = ws_V + (size_t)Bc * Nc * Ec;          // B*E
    float* ws_logits = ws_ctx + Bc * Ec;                     // B*E

    qproj_kernel<<<dim3(Bc), dim3(Ec), 0, stream>>>(ds, Wq, bq, ws_q);
    kv_att_kernel<<<dim3(Bc, TGROUPS), dim3(256), 0, stream>>>(
        ne, Wk, bk, Wv, bv, mask, ws_q, ws_att, ws_V);
    softmax_ctx_kernel<<<dim3(Bc * Hc), dim3(256), 0, stream>>>(ws_att, ws_V, ws_ctx);
    out_ln_proj_kernel<<<dim3(Bc), dim3(Ec), 0, stream>>>(
        ws_ctx, Wo, bo, Wp, bp, lng, lnb, ws_logits);
    scores_kernel<<<dim3(Bc, (Nc + 255) / 256), dim3(256), 0, stream>>>(
        ws_logits, ne, visited, prev, allv, ew, sb, rp, out);
}
